// VccT5LayerSelfAttention_36051955483037
// MI455X (gfx1250) — compile-verified, matrix operand order checked by reading, not run on hardware
//
#include <hip/hip_runtime.h>
#include <math.h>

// Problem constants
#define BB    2
#define SS    2048
#define DM    1024
#define NH    16
#define DKV   64
#define NTOK  (BB * SS)          // 4096
#define INNER (NH * DKV)         // 1024

typedef __bf16 bf16;
typedef __attribute__((ext_vector_type(16))) __bf16 v16bf;
typedef __attribute__((ext_vector_type(8)))  float  v8f;

typedef unsigned uint32x4 __attribute__((ext_vector_type(4)));
typedef int      int32x8  __attribute__((ext_vector_type(8)));
typedef int      int32x4  __attribute__((ext_vector_type(4)));

// address-space qualified int4 for the async-to-LDS builtin
typedef __attribute__((address_space(1))) int32x4 as1_int4;
typedef __attribute__((address_space(3))) int32x4 as3_int4;

union FragAB { v16bf v; uint4 u[2]; };
struct alignas(8) bf16x4 { bf16 v[4]; };

#if __has_include(<hip/amd_detail/amd_gfx1250_TDM.h>)
#define TDM_6ARG 1
#endif

// ---------------- workspace layout (bytes) ----------------
#define OFF_NORMED 0UL                         // 4096*1024 bf16  = 8 MiB
#define OFF_WT     (OFF_NORMED + 8388608UL)    // 4 * 1024*1024 bf16 = 8 MiB
#define OFF_Q      (OFF_WT     + 8388608UL)    // 8 MiB  [B,H,S,64] bf16
#define OFF_K      (OFF_Q      + 8388608UL)    // 8 MiB  [B,H,S,64] bf16
#define OFF_VT     (OFF_K      + 8388608UL)    // 8 MiB  [B,H,64,S] bf16
#define OFF_CTX    (OFF_VT     + 8388608UL)    // 8 MiB  [B,S,1024] bf16
#define OFF_KBIAS  (OFF_CTX    + 8388608UL)    // 16 KiB [B,S] f32

// ---------------- CDNA5 data-movement helpers ----------------
__device__ __forceinline__ void wait_async_all() {
#if __has_builtin(__builtin_amdgcn_s_wait_asynccnt)
    __builtin_amdgcn_s_wait_asynccnt(0);
#else
    asm volatile("s_wait_asynccnt 0" ::: "memory");
#endif
}
__device__ __forceinline__ void wait_tensor_all() {
#if __has_builtin(__builtin_amdgcn_s_wait_tensorcnt)
    __builtin_amdgcn_s_wait_tensorcnt(0);
#else
    asm volatile("s_wait_tensorcnt 0" ::: "memory");
#endif
}

// 16-byte global -> LDS copy; async (ASYNCcnt) when available.
__device__ __forceinline__ void copy16(bf16* l, const bf16* g) {
#if __has_builtin(__builtin_amdgcn_global_load_async_to_lds_b128)
    __builtin_amdgcn_global_load_async_to_lds_b128(
        (as1_int4*)g, (as3_int4*)l, 0, 0);
#else
    *(uint4*)l = *(const uint4*)g;
#endif
}

#if __has_builtin(__builtin_amdgcn_tensor_load_to_lds)
#define HAVE_TDM 1
// TDM: load a 64(row) x 32(elem) bf16 tile from a [1024][1024] bf16 tensor
// into LDS with 8-bf16 padding per row (pad_interval=16 dwords, pad_amount=4
// dwords) so LDS row stride is 40 bf16 = 80B (conflict-free for b128 reads).
__device__ __forceinline__ void tdm_load_b_tile(const bf16* gtile, unsigned lds_byte_off)
{
    unsigned long long ga = (unsigned long long)(size_t)gtile;
    uint32x4 g0 = { 1u,                                   // count=1 (valid D#)
                    lds_byte_off,                          // lds_addr
                    (unsigned)(ga & 0xffffffffu),          // global_addr[31:0]
                    (unsigned)((ga >> 32) & 0x01ffffffu) | (2u << 30) }; // [56:32], type=2
    int32x8 g1 = { (int)((1u << 16)                        // data_size=1 -> 2 bytes
                       | (1u << 20)                        // pad_enable
                       | (3u << 22)                        // pad_interval: 16 dwords
                       | (3u << 25)),                      // pad_amount: 4 dwords
                   (int)(1024u << 16),                     // tensor_dim0=1024 (lo16)
                   (int)(1024u << 16),                     // dim0 hi | tensor_dim1=1024 (lo16)
                   (int)(32u << 16),                       // dim1 hi | tile_dim0=32
                   64,                                     // tile_dim1=64, tile_dim2=0
                   1024,                                   // tensor_dim0_stride[31:0]
                   0, 0 };
    int32x4 gz = { 0, 0, 0, 0 };
#ifdef TDM_6ARG
    int32x8 gz8 = { 0, 0, 0, 0, 0, 0, 0, 0 };
    __builtin_amdgcn_tensor_load_to_lds(g0, g1, gz, gz, gz8, 0);
#else
    __builtin_amdgcn_tensor_load_to_lds(g0, g1, gz, gz, 0);
#endif
}
#endif

// ---------------- 1) RMSNorm -> bf16 ----------------
__global__ __launch_bounds__(256) void rmsnorm_kernel(
    const float* __restrict__ x, const float* __restrict__ w, bf16* __restrict__ out)
{
    __shared__ float sWave[8];
    const int row = blockIdx.x, tid = threadIdx.x;
    const float4 xv = ((const float4*)(x + (size_t)row * DM))[tid];
    float ss = xv.x * xv.x + xv.y * xv.y + xv.z * xv.z + xv.w * xv.w;
#pragma unroll
    for (int off = 16; off; off >>= 1) ss += __shfl_down(ss, off, 32);
    if ((tid & 31) == 0) sWave[tid >> 5] = ss;
    __syncthreads();
    if (tid == 0) {
        float t = 0.f;
#pragma unroll
        for (int i = 0; i < 8; ++i) t += sWave[i];
        sWave[0] = rsqrtf(t * (1.0f / DM) + 1e-6f);
    }
    __syncthreads();
    const float sc = sWave[0];
    const float4 wv = ((const float4*)w)[tid];
    bf16x4 o;
    o.v[0] = (bf16)(xv.x * sc * wv.x);
    o.v[1] = (bf16)(xv.y * sc * wv.y);
    o.v[2] = (bf16)(xv.z * sc * wv.z);
    o.v[3] = (bf16)(xv.w * sc * wv.w);
    ((bf16x4*)(out + (size_t)row * DM))[tid] = o;
}

// ---------------- 2) weight transpose + bf16 convert ----------------
__global__ __launch_bounds__(256) void wconv_kernel(
    const float* __restrict__ W0, const float* __restrict__ W1,
    const float* __restrict__ W2, const float* __restrict__ W3,
    bf16* __restrict__ dst)
{
    __shared__ float tile[32][33];
    const float* src = (blockIdx.z == 0) ? W0 : (blockIdx.z == 1) ? W1
                     : (blockIdx.z == 2) ? W2 : W3;
    bf16* out = dst + (size_t)blockIdx.z * DM * INNER;
    const int bx = blockIdx.x * 32, by = blockIdx.y * 32;
    const int tx = threadIdx.x, ty = threadIdx.y;
#pragma unroll
    for (int j = 0; j < 32; j += 8)
        tile[ty + j][tx] = src[(size_t)(by + ty + j) * INNER + bx + tx];
    __syncthreads();
#pragma unroll
    for (int j = 0; j < 32; j += 8)
        out[(size_t)(bx + ty + j) * DM + by + tx] = (bf16)tile[tx][ty + j];
}

// ---------------- 3) per-key additive bias ----------------
__global__ __launch_bounds__(256) void kbias_kernel(
    const float* __restrict__ mask, const float* __restrict__ score,
    const int* __restrict__ dups, float* __restrict__ out)
{
    const int i = blockIdx.x * 256 + threadIdx.x;
    out[i] = -1000.f * (1.f - mask[i]) + logf(score[i] + 1e-4f)
           + logf((float)dups[i] + 1e-4f);
}

// ---------------- 4) bf16 GEMM: C[4096,1024] = A[4096,1024] * BT^T ----------------
// mode 0: store Q [B,H,S,64] bf16 ; mode 1: K same ; mode 2: V transposed [B,H,64,S]
// mode 3: store f32 to out[M,1024]
__device__ __forceinline__ void store_c(int mode, void* out, int row, int col, float v)
{
    if (mode == 3) {
        ((float*)out)[(size_t)row * DM + col] = v;
    } else {
        const int b = row >> 11, s = row & 2047;
        const int hh = col >> 6, d = col & 63;
        bf16* o = (bf16*)out;
        if (mode == 2)
            o[(((size_t)(b * NH + hh)) * DKV + d) * SS + s] = (bf16)v;
        else
            o[(((size_t)(b * NH + hh)) * SS + s) * DKV + d] = (bf16)v;
    }
}

__global__ __launch_bounds__(256) void gemm_bf16_kernel(
    const bf16* __restrict__ A, const bf16* __restrict__ BT,
    int mode, void* __restrict__ out)
{
    __shared__ __align__(16) bf16 sA[256][40];   // 20.0 KiB, padded rows
    __shared__ __align__(16) bf16 sB[64][40];    //  5.0 KiB, padded rows
    const int tid = threadIdx.x, w = tid >> 5, l = tid & 31;
    const int lh = l >> 4, ln = l & 15, c0 = lh * 8;
    const int mBase = blockIdx.x * 256, nBase = blockIdx.y * 64;

    v8f acc0[4] = {}, acc1[4] = {};

    for (int k0 = 0; k0 < DM; k0 += 32) {
        __syncthreads();
        // A tile: 256 x 32 bf16 -> async copies, 4x16B per thread
#pragma unroll
        for (int j = 0; j < 4; ++j) {
            const int chunk = tid + j * 256;          // 1024 chunks of 8 bf16
            const int r = chunk >> 2, cc = (chunk & 3) * 8;
            copy16(&sA[r][cc], A + (size_t)(mBase + r) * DM + k0 + cc);
        }
#ifdef HAVE_TDM
        if (w == 0)  // one TDM descriptor moves the whole B tile (with LDS padding)
            tdm_load_b_tile(BT + (size_t)nBase * DM + k0, (unsigned)(size_t)(void*)sB);
#else
        {
            const int r = tid >> 2, cc = (tid & 3) * 8;
            copy16(&sB[r][cc], BT + (size_t)(nBase + r) * DM + k0 + cc);
        }
#endif
        wait_async_all();
#ifdef HAVE_TDM
        if (w == 0) wait_tensor_all();
#endif
        __syncthreads();

        FragAB a0, a1;
        a0.u[0] = *(const uint4*)&sA[w * 16 + ln][c0];
        a0.u[1] = *(const uint4*)&sA[w * 16 + ln][c0 + 16];
        a1.u[0] = *(const uint4*)&sA[128 + w * 16 + ln][c0];
        a1.u[1] = *(const uint4*)&sA[128 + w * 16 + ln][c0 + 16];
#pragma unroll
        for (int g = 0; g < 4; ++g) {
            FragAB b;
            b.u[0] = *(const uint4*)&sB[g * 16 + ln][c0];
            b.u[1] = *(const uint4*)&sB[g * 16 + ln][c0 + 16];
            acc0[g] = __builtin_amdgcn_wmma_f32_16x16x32_bf16(
                false, a0.v, false, b.v, (short)0, acc0[g], false, false);
            acc1[g] = __builtin_amdgcn_wmma_f32_16x16x32_bf16(
                false, a1.v, false, b.v, (short)0, acc1[g], false, false);
        }
    }

#pragma unroll
    for (int g = 0; g < 4; ++g) {
#pragma unroll
        for (int i = 0; i < 8; ++i) {
            const int col = nBase + g * 16 + ln;
            store_c(mode, out, mBase + w * 16 + lh * 8 + i, col, acc0[g][i]);
            store_c(mode, out, mBase + 128 + w * 16 + lh * 8 + i, col, acc1[g][i]);
        }
    }
}

// ---------------- 5) relative-position bucket ----------------
__device__ __forceinline__ int rel_bucket(int rel)
{
    int ret = (rel > 0) ? 16 : 0;
    int n = (rel < 0) ? -rel : rel;
    if (n < 8) return ret + n;
    // 8 + log(n/8)/log(16) * 8 ; 8/log(16) = 2.8853900817779268
    int v = 8 + (int)(logf((float)n * 0.125f) * 2.8853900817779268f);
    v = (v < 15) ? v : 15;
    return ret + v;
}

// ---------------- 6) flash attention ----------------
__global__ __launch_bounds__(256) void attn_kernel(
    const bf16* __restrict__ Q, const bf16* __restrict__ K,
    const bf16* __restrict__ Vt, bf16* __restrict__ Ctx,
    const int* __restrict__ positions, const float* __restrict__ kbias,
    const float* __restrict__ rel_emb)
{
    __shared__ float sRel[32];
    __shared__ __align__(16) bf16 sP[8][16][72];

    const int b = blockIdx.z, h = blockIdx.y;
    const int qBase = blockIdx.x * 128;
    const int tid = threadIdx.x, w = tid >> 5, l = tid & 31;
    const int lh = l >> 4, ln = l & 15, c0 = lh * 8;

    if (tid < 32) sRel[tid] = rel_emb[tid * NH + h];
    __syncthreads();

    const bf16* Qbh = Q  + ((size_t)(b * NH + h)) * SS * DKV;
    const bf16* Kbh = K  + ((size_t)(b * NH + h)) * SS * DKV;
    const bf16* Vbh = Vt + ((size_t)(b * NH + h)) * DKV * SS;
    const int*  posb = positions + b * SS;
    const float* kbb = kbias + b * SS;

    // Q fragments (persistent): rows w*16+ln, d split per lane-half
    FragAB aQ[2];
    const int qrow = qBase + w * 16 + ln;
#pragma unroll
    for (int s = 0; s < 2; ++s) {
        aQ[s].u[0] = *(const uint4*)(Qbh + (size_t)qrow * DKV + s * 32 + c0);
        aQ[s].u[1] = *(const uint4*)(Qbh + (size_t)qrow * DKV + s * 32 + c0 + 16);
    }
    int posq[8];
#pragma unroll
    for (int i = 0; i < 8; ++i) posq[i] = posb[qBase + w * 16 + lh * 8 + i];

    float mMax[8], lSum[8];
#pragma unroll
    for (int i = 0; i < 8; ++i) { mMax[i] = -3.0e38f; lSum[i] = 0.f; }
    v8f acc[4] = {};

    for (int kt = 0; kt < SS / 64; ++kt) {
        const int kBase = kt * 64;
        float p[4][8];
        float tileMax[8];
#pragma unroll
        for (int i = 0; i < 8; ++i) tileMax[i] = -3.0e38f;

        // scores for 64 keys
#pragma unroll
        for (int g = 0; g < 4; ++g) {
            const int krow = kBase + g * 16 + ln;
            v8f sAcc = {};
#pragma unroll
            for (int s = 0; s < 2; ++s) {
                FragAB bK;
                bK.u[0] = *(const uint4*)(Kbh + (size_t)krow * DKV + s * 32 + c0);
                bK.u[1] = *(const uint4*)(Kbh + (size_t)krow * DKV + s * 32 + c0 + 16);
                sAcc = __builtin_amdgcn_wmma_f32_16x16x32_bf16(
                    false, aQ[s].v, false, bK.v, (short)0, sAcc, false, false);
            }
            const float kb = kbb[krow];
            const int   pk = posb[krow];
#pragma unroll
            for (int i = 0; i < 8; ++i) {
                const float v = sAcc[i] + sRel[rel_bucket(pk - posq[i])] + kb;
                p[g][i] = v;
                tileMax[i] = fmaxf(tileMax[i], v);
            }
        }
        // row max across the 16-lane half
#pragma unroll
        for (int off = 1; off < 16; off <<= 1)
#pragma unroll
            for (int i = 0; i < 8; ++i)
                tileMax[i] = fmaxf(tileMax[i], __shfl_xor(tileMax[i], off, 32));

        float scale[8], rowSum[8];
#pragma unroll
        for (int i = 0; i < 8; ++i) {
            const float nm = fmaxf(mMax[i], tileMax[i]);
            scale[i] = __expf(mMax[i] - nm);
            mMax[i] = nm;
            rowSum[i] = 0.f;
        }
#pragma unroll
        for (int g = 0; g < 4; ++g)
#pragma unroll
            for (int i = 0; i < 8; ++i) {
                p[g][i] = __expf(p[g][i] - mMax[i]);
                rowSum[i] += p[g][i];
            }
#pragma unroll
        for (int off = 1; off < 16; off <<= 1)
#pragma unroll
            for (int i = 0; i < 8; ++i)
                rowSum[i] += __shfl_xor(rowSum[i], off, 32);
#pragma unroll
        for (int i = 0; i < 8; ++i) lSum[i] = lSum[i] * scale[i] + rowSum[i];
#pragma unroll
        for (int g = 0; g < 4; ++g)
#pragma unroll
            for (int i = 0; i < 8; ++i) acc[g][i] *= scale[i];

        // C-layout -> A-layout via per-wave LDS round trip
#pragma unroll
        for (int g = 0; g < 4; ++g)
#pragma unroll
            for (int i = 0; i < 8; ++i)
                sP[w][lh * 8 + i][g * 16 + ln] = (bf16)p[g][i];
        asm volatile("s_wait_dscnt 0" ::: "memory");

        FragAB aP[2];
#pragma unroll
        for (int s = 0; s < 2; ++s) {
            aP[s].u[0] = *(const uint4*)&sP[w][ln][s * 32 + c0];
            aP[s].u[1] = *(const uint4*)&sP[w][ln][s * 32 + c0 + 16];
        }
        asm volatile("s_wait_dscnt 0" ::: "memory");

        // ctx += P * V  (B operand from transposed V, contiguous rows)
#pragma unroll
        for (int g = 0; g < 4; ++g) {
            const int drow = g * 16 + ln;
#pragma unroll
            for (int s = 0; s < 2; ++s) {
                FragAB bV;
                bV.u[0] = *(const uint4*)(Vbh + (size_t)drow * SS + kBase + s * 32 + c0);
                bV.u[1] = *(const uint4*)(Vbh + (size_t)drow * SS + kBase + s * 32 + c0 + 16);
                acc[g] = __builtin_amdgcn_wmma_f32_16x16x32_bf16(
                    false, aP[s].v, false, bV.v, (short)0, acc[g], false, false);
            }
        }
    }

    // normalize + store ctx as bf16 [B,S,INNER]
#pragma unroll
    for (int i = 0; i < 8; ++i) lSum[i] = 1.0f / lSum[i];
#pragma unroll
    for (int g = 0; g < 4; ++g)
#pragma unroll
        for (int i = 0; i < 8; ++i) {
            const int row = qBase + w * 16 + lh * 8 + i;
            const int col = h * DKV + g * 16 + ln;
            Ctx[((size_t)(b * SS + row)) * INNER + col] = (bf16)(acc[g][i] * lSum[i]);
        }
}

// ---------------- host launcher ----------------
extern "C" void kernel_launch(void* const* d_in, const int* in_sizes, int n_in,
                              void* d_out, int out_size, void* d_ws, size_t ws_size,
                              hipStream_t stream)
{
    const float* hidden    = (const float*)d_in[0];
    const int*   positions = (const int*)  d_in[1];
    const float* amask     = (const float*)d_in[2];
    const float* sscores   = (const float*)d_in[3];
    const int*   ndups     = (const int*)  d_in[4];
    const float* lnw       = (const float*)d_in[5];
    const float* Wq        = (const float*)d_in[6];
    const float* Wk        = (const float*)d_in[7];
    const float* Wv        = (const float*)d_in[8];
    const float* Wo        = (const float*)d_in[9];
    const float* rel_emb   = (const float*)d_in[10];

    char* ws = (char*)d_ws;
    bf16* normed = (bf16*)(ws + OFF_NORMED);
    bf16* wT     = (bf16*)(ws + OFF_WT);      // [4][1024][1024] : WqT,WkT,WvT,WoT
    bf16* qws    = (bf16*)(ws + OFF_Q);
    bf16* kws    = (bf16*)(ws + OFF_K);
    bf16* vtws   = (bf16*)(ws + OFF_VT);
    bf16* ctxws  = (bf16*)(ws + OFF_CTX);
    float* kbws  = (float*)(ws + OFF_KBIAS);

    rmsnorm_kernel<<<NTOK, 256, 0, stream>>>(hidden, lnw, normed);
    wconv_kernel<<<dim3(32, 32, 4), dim3(32, 8), 0, stream>>>(Wq, Wk, Wv, Wo, wT);
    kbias_kernel<<<NTOK / 256, 256, 0, stream>>>(amask, sscores, ndups, kbws);

    const dim3 gg(NTOK / 256, INNER / 64, 1);
    gemm_bf16_kernel<<<gg, 256, 0, stream>>>(normed, wT + 0UL * DM * INNER, 0, qws);
    gemm_bf16_kernel<<<gg, 256, 0, stream>>>(normed, wT + 1UL * DM * INNER, 1, kws);
    gemm_bf16_kernel<<<gg, 256, 0, stream>>>(normed, wT + 2UL * DM * INNER, 2, vtws);

    attn_kernel<<<dim3(SS / 128, NH, BB), 256, 0, stream>>>(
        qws, kws, vtws, ctxws, positions, kbws, rel_emb);

    gemm_bf16_kernel<<<gg, 256, 0, stream>>>(ctxws, wT + 3UL * DM * INNER, 3, d_out);
}
